// Heatmap2dMaxDetBlock_28424093565179
// MI455X (gfx1250) — compile-verified, matrix operand order checked by reading, not run on hardware
//
#include <hip/hip_runtime.h>
#include <hip/hip_bf16.h>
#include <math.h>

#define HH 512
#define WW 512
#define KK 64
#define NPTS 10
#define MIN_PAF 0.05f
#define PEAK_THR 0.1f
#define NMS_DIST2 36
#define NMAPS 152            // 8 images * 19 channels (also 8 * 19 limbs)
#define NBINS 4096
#define CANDMAX 2048

typedef _Float16 v16h __attribute__((ext_vector_type(16)));
typedef float    v8f  __attribute__((ext_vector_type(8)));

#define HAS_ASYNC_LDS __has_builtin(__builtin_amdgcn_global_load_async_to_lds_b32)

typedef __attribute__((address_space(1))) int glb_i32;
typedef __attribute__((address_space(3))) int lds_i32;

__constant__ int c_kpt[19][2] = {
  {1,2},{1,5},{2,3},{3,4},{5,6},{6,7},{1,8},{8,9},{9,10},{1,11},
  {11,12},{12,13},{1,0},{0,14},{14,16},{0,15},{15,17},{2,16},{5,17}};
__constant__ int c_paf[19][2] = {
  {12,13},{20,21},{14,15},{16,17},{22,23},{24,25},{0,1},{2,3},{4,5},{6,7},
  {8,9},{10,11},{28,29},{30,31},{34,35},{32,33},{36,37},{18,19},{26,27}};

__device__ __forceinline__ float thf(float v) { return v < PEAK_THR ? 0.0f : v; }

// Thresholded 4-neighbor local-max test (zero padding at borders, like jnp.pad).
__device__ __forceinline__ bool is_peak(const float* b, int x, int y, float& cv) {
  float c = thf(b[y * WW + x]);
  cv = c;
  if (c <= 0.0f) return false;               // c==0 can never be > neighbors (>=0)
  float r = (x < WW-1) ? thf(b[y * WW + x + 1]) : 0.0f;
  float l = (x > 0)    ? thf(b[y * WW + x - 1]) : 0.0f;
  float d = (y < HH-1) ? thf(b[(y + 1) * WW + x]) : 0.0f;
  float u = (y > 0)    ? thf(b[(y - 1) * WW + x]) : 0.0f;
  return (c > r) & (c > l) & (c > d) & (c > u);
}

// ---------------------------------------------------------------------------
// Kernel 1: per (image,channel) map -> top-64 peaks via 2-level radix-select,
// x-sort, greedy NMS. One 256-thread block per map. Writes coords/ss/ks and
// stashes peak arrays in workspace for the limb kernel.
// ---------------------------------------------------------------------------
__global__ void __launch_bounds__(256)
peak_topk_nms_kernel(const float* __restrict__ hm,
                     float* __restrict__ out_coords,
                     float* __restrict__ out_ss,
                     float* __restrict__ out_ks,
                     int* __restrict__ wpx, int* __restrict__ wpy,
                     float* __restrict__ wps, int* __restrict__ wpk) {
  __shared__ int   hist[NBINS];
  __shared__ float cval[CANDMAX];
  __shared__ int   cx[CANDMAX];
  __shared__ int   cy[CANDMAX];
  __shared__ int   ccount;
  __shared__ int   s_cut1, s_above1, s_cut2;
  __shared__ float rv[256];
  __shared__ int   ri[256];
  __shared__ float selv[KK];
  __shared__ int   selx[KK], sely[KK], selk[KK];

  const int m   = blockIdx.x;
  const int tid = threadIdx.x;
  const float* b = hm + (size_t)m * (HH * WW);

  // ---- pass 1: level-1 histogram of float bits [30:19] over peaks ----
  for (int i = tid; i < NBINS; i += 256) hist[i] = 0;
  __syncthreads();
  for (int idx = tid; idx < HH * WW; idx += 256) {
    __builtin_prefetch(b + ((idx + 4096) & (HH * WW - 1)), 0, 1);
    float c;
    if (is_peak(b, idx & (WW - 1), idx >> 9, c)) {
      int bin = (__float_as_uint(c) >> 19) & (NBINS - 1);
      atomicAdd(&hist[bin], 1);
    }
  }
  __syncthreads();
  if (tid == 0) {
    int acc = 0, cut = 0, above = 0;
    for (int bn = NBINS - 1; bn >= 0; --bn) {
      int h = hist[bn];
      if (acc + h >= KK) { cut = bn; above = acc; break; }
      acc += h;
    }
    s_cut1 = cut; s_above1 = above;
  }
  __syncthreads();
  const int cut1 = s_cut1;

  // ---- pass 2: level-2 histogram (bits [18:7]) inside the cut bin ----
  for (int i = tid; i < NBINS; i += 256) hist[i] = 0;
  __syncthreads();
  for (int idx = tid; idx < HH * WW; idx += 256) {
    float c;
    if (is_peak(b, idx & (WW - 1), idx >> 9, c)) {
      unsigned u = __float_as_uint(c);
      if (((u >> 19) & (NBINS - 1)) == (unsigned)cut1)
        atomicAdd(&hist[(u >> 7) & (NBINS - 1)], 1);
    }
  }
  __syncthreads();
  if (tid == 0) {
    int acc = s_above1, cut = 0;
    for (int bn = NBINS - 1; bn >= 0; --bn) {
      acc += hist[bn];
      if (acc >= KK) { cut = bn; break; }
    }
    s_cut2 = cut;
    ccount = 0;
  }
  __syncthreads();
  const int cut2 = s_cut2;

  // ---- pass 3: compact candidates above the 24-bit cut into LDS ----
  for (int idx = tid; idx < HH * WW; idx += 256) {
    float c;
    int x = idx & (WW - 1), y = idx >> 9;
    if (is_peak(b, x, y, c)) {
      unsigned u  = __float_as_uint(c);
      int bin1 = (u >> 19) & (NBINS - 1);
      int bin2 = (u >> 7)  & (NBINS - 1);
      if (bin1 > cut1 || (bin1 == cut1 && bin2 >= cut2)) {
        int pos = atomicAdd(&ccount, 1);
        if (pos < CANDMAX) { cval[pos] = c; cx[pos] = x; cy[pos] = y; }
      }
    }
  }
  __syncthreads();
  const int C = min(ccount, CANDMAX);

  // ---- top-64 extraction (candidate list is tiny, ~100 entries) ----
  for (int k = 0; k < KK; ++k) {
    float bv = -1.0f; int bi = 0x7FFFFFFF;
    for (int i = tid; i < C; i += 256) {
      float v = cval[i];
      if (v > bv || (v == bv && i < bi)) { bv = v; bi = i; }
    }
    rv[tid] = bv; ri[tid] = bi;
    __syncthreads();
    for (int s = 128; s > 0; s >>= 1) {
      if (tid < s) {
        if (rv[tid + s] > rv[tid] ||
            (rv[tid + s] == rv[tid] && ri[tid + s] < ri[tid])) {
          rv[tid] = rv[tid + s]; ri[tid] = ri[tid + s];
        }
      }
      __syncthreads();
    }
    if (tid == 0) {
      if (rv[0] > 0.0f) {
        int w = ri[0];
        selv[k] = rv[0]; selx[k] = cx[w]; sely[k] = cy[w]; selk[k] = 1;
        cval[w] = -1.0f;
      } else {
        selv[k] = 0.0f; selx[k] = 0; sely[k] = 0; selk[k] = 0;
      }
    }
    __syncthreads();
  }

  // ---- x-sort (stable insertion, invalid last) + greedy NMS ----
  if (tid == 0) {
    for (int i = 1; i < KK; ++i) {
      float v = selv[i]; int x = selx[i], y = sely[i], kv = selk[i];
      int key = kv ? x : (WW + 1);
      int j = i - 1;
      while (j >= 0 && (selk[j] ? selx[j] : (WW + 1)) > key) {
        selv[j+1] = selv[j]; selx[j+1] = selx[j];
        sely[j+1] = sely[j]; selk[j+1] = selk[j]; --j;
      }
      selv[j+1] = v; selx[j+1] = x; sely[j+1] = y; selk[j+1] = kv;
    }
    for (int i = 0; i < KK; ++i) {
      if (!selk[i]) continue;
      for (int j = i + 1; j < KK; ++j) {
        if (!selk[j]) continue;
        int dx = selx[j] - selx[i], dy = sely[j] - sely[i];
        if (dx * dx + dy * dy < NMS_DIST2) selk[j] = 0;
      }
    }
  }
  __syncthreads();

  // ---- outputs + workspace for limb kernel ----
  if (tid < KK) {
    size_t o = (size_t)m * KK + tid;
    out_coords[o * 2 + 0] = (float)selx[tid] * 2.0f;
    out_coords[o * 2 + 1] = (float)sely[tid] * 2.0f;
    out_ss[o] = selv[tid];
    out_ks[o] = selk[tid] ? 1.0f : 0.0f;
    wpx[o] = selx[tid]; wpy[o] = sely[tid]; wps[o] = selv[tid]; wpk[o] = selk[tid];
  }
}

// ---------------------------------------------------------------------------
// Kernel 2: per (image,limb) 64x64 pairwise limb scores. dx/dy pair-geometry
// matrices are rank-2 outer products computed on the WMMA unit
// (v_wmma_f32_16x16x32_f16, exact for integer coords <= 511); the WMMA C/D
// lane layout then hands each lane its 8 (i,j) pairs for PAF sampling.
// Peak arrays are staged L2 -> LDS via gfx1250 async-to-LDS when available.
// ---------------------------------------------------------------------------
__global__ void __launch_bounds__(256)
limb_score_kernel(const float* __restrict__ paf,
                  const int* __restrict__ wbase,   // ws base: [px|py|ps|pk] blocks
                  float* __restrict__ out_conn,
                  float* __restrict__ out_cvalid) {
  // raw staging: 8 segments of 64 dwords: a:{px,py,ps,pk} then b:{px,py,ps,pk}
  __shared__ int   s_raw[8 * KK];
  __shared__ float s_xa[KK], s_ya[KK], s_xb[KK], s_yb[KK];

  const int n = blockIdx.x / 19;   // image
  const int l = blockIdx.x % 19;   // limb
  const int tid = threadIdx.x;
  const int ma = n * 19 + c_kpt[l][0];
  const int mb = n * 19 + c_kpt[l][1];

  // ---- stage 8 x 64 dwords of peak data into LDS ----
#if HAS_ASYNC_LDS
  for (int s = tid; s < 8 * KK; s += 256) {
    int a = s >> 6, k = s & 63;
    int arr = a & 3, mm = (a < 4) ? ma : mb;
    const int* g = wbase + (size_t)arr * (NMAPS * KK) + (size_t)mm * KK + k;
    __builtin_amdgcn_global_load_async_to_lds_b32(
        (glb_i32*)(void*)const_cast<int*>(g),
        (lds_i32*)(void*)&s_raw[s], 0, 0);
  }
#if __has_builtin(__builtin_amdgcn_s_wait_asynccnt)
  __builtin_amdgcn_s_wait_asynccnt(0);
#else
  asm volatile("s_wait_asynccnt 0x0" ::: "memory");
#endif
#else
  for (int s = tid; s < 8 * KK; s += 256) {
    int a = s >> 6, k = s & 63;
    int arr = a & 3, mm = (a < 4) ? ma : mb;
    s_raw[s] = wbase[(size_t)arr * (NMAPS * KK) + (size_t)mm * KK + k];
  }
#endif
  __syncthreads();

  // typed views + int->float conversion for coordinates
  const int*   pxa = s_raw + 0 * KK;
  const int*   pya = s_raw + 1 * KK;
  const float* psa = (const float*)(s_raw + 2 * KK);
  const int*   pka = s_raw + 3 * KK;
  const int*   pxb = s_raw + 4 * KK;
  const int*   pyb = s_raw + 5 * KK;
  const float* psb = (const float*)(s_raw + 6 * KK);
  const int*   pkb = s_raw + 7 * KK;
  if (tid < KK) {
    s_xa[tid] = (float)pxa[tid]; s_ya[tid] = (float)pya[tid];
    s_xb[tid] = (float)pxb[tid]; s_yb[tid] = (float)pyb[tid];
  }
  __syncthreads();

  const float* pfx = paf + (size_t)(n * 38 + c_paf[l][0]) * (HH * WW);
  const float* pfy = paf + (size_t)(n * 38 + c_paf[l][1]) * (HH * WW);

  const int wave = tid >> 5;
  const int lane = tid & 31;
  const int grp  = lane >> 4;      // lane group (0: rows 0-7, 1: rows 8-15)
  const int l16  = lane & 15;
  const size_t obase = (size_t)blockIdx.x * KK * KK;

  for (int t = wave; t < 16; t += 8) {       // 16 tiles of 16x16 pairs
    const int ti = t >> 2, tj = t & 3;

    // A[i,:] = (-a_i, 1, 0...), B[:,j] = (1, b_j, 0...)^T  =>  D[i,j] = b_j - a_i
    v16h aX = {}, bX = {}, aY = {}, bY = {};
    if (grp == 0) {
      float ax = s_xa[ti * 16 + l16], ay = s_ya[ti * 16 + l16];
      float bx = s_xb[tj * 16 + l16], by = s_yb[tj * 16 + l16];
      aX[0] = (_Float16)(-ax); aX[1] = (_Float16)1.0f;
      aY[0] = (_Float16)(-ay); aY[1] = (_Float16)1.0f;
      bX[0] = (_Float16)1.0f;  bX[1] = (_Float16)bx;
      bY[0] = (_Float16)1.0f;  bY[1] = (_Float16)by;
    }
    v8f dxm = {}, dym = {};
    dxm = __builtin_amdgcn_wmma_f32_16x16x32_f16(false, aX, false, bX,
                                                 (short)0, dxm, false, false);
    dym = __builtin_amdgcn_wmma_f32_16x16x32_f16(false, aY, false, bY,
                                                 (short)0, dym, false, false);

    const int   j   = tj * 16 + l16;         // D layout: lane -> column
    const int   kbj = pkb[j];                // loop-invariant per lane
    const float sbj = psb[j];

#pragma unroll
    for (int v = 0; v < 8; ++v) {            // VGPR v -> row v + 8*grp
      const int i = ti * 16 + v + (grp << 3);
      float dx = dxm[v], dy = dym[v];
      float norm = sqrtf(dx * dx + dy * dy);
      float safe = fmaxf(norm, 1e-6f);
      float ux = dx / safe, uy = dy / safe;
      float ax = s_xa[i], ay = s_ya[i];

      int npass = 0; float ssum = 0.0f;
#pragma unroll
      for (int p = 0; p < NPTS; ++p) {
        float tt = (float)p * (1.0f / (NPTS - 1));
        int sx = (int)rintf(ax + tt * dx); sx = min(max(sx, 0), WW - 1);
        int sy = (int)rintf(ay + tt * dy); sy = min(max(sy, 0), HH - 1);
        float fx = pfx[sy * WW + sx];
        float fy = pfy[sy * WW + sx];
        float sc = ux * fx + uy * fy;
        bool pass = sc > MIN_PAF;
        npass += pass;
        ssum  += pass ? sc : 0.0f;
      }
      float succ  = (float)npass * (1.0f / NPTS);
      float ratio = (npass > 0) ? (ssum / (float)npass) : 0.0f;
      ratio += fminf((float)(HH / 2) / safe - 1.0f, 0.0f);
      bool valid = (ratio > 0.0f) && (succ > 0.8f) &&
                   (pka[i] != 0) && (kbj != 0) && (norm > 0.0f);
      float score = valid ? (ratio + psa[i] + sbj) : 0.0f;

      out_conn[obase + (size_t)i * KK + j]   = score;
      out_cvalid[obase + (size_t)i * KK + j] = valid ? 1.0f : 0.0f;
    }
  }
}

extern "C" void kernel_launch(void* const* d_in, const int* in_sizes, int n_in,
                              void* d_out, int out_size, void* d_ws, size_t ws_size,
                              hipStream_t stream) {
  const float* hm  = (const float*)d_in[0];   // (8,19,512,512)
  const float* paf = (const float*)d_in[1];   // (8,38,512,512)

  float* out = (float*)d_out;
  const size_t n_coords = (size_t)NMAPS * KK * 2;   // 19456
  const size_t n_ss     = (size_t)NMAPS * KK;       // 9728
  float* out_coords = out;
  float* out_ss     = out_coords + n_coords;
  float* out_ks     = out_ss + n_ss;
  float* out_conn   = out_ks + n_ss;
  float* out_cvalid = out_conn + (size_t)NMAPS * KK * KK;

  const size_t npk = (size_t)NMAPS * KK;
  int*   wpx = (int*)d_ws;                 // ws layout: [px | py | ps | pk]
  int*   wpy = wpx + npk;
  float* wps = (float*)(wpy + npk);
  int*   wpk = (int*)(wps + npk);

  peak_topk_nms_kernel<<<NMAPS, 256, 0, stream>>>(
      hm, out_coords, out_ss, out_ks, wpx, wpy, wps, wpk);
  limb_score_kernel<<<NMAPS, 256, 0, stream>>>(
      paf, wpx, out_conn, out_cvalid);
}
